// AttentionBlock_16509854285862
// MI455X (gfx1250) — matrix operand order repaired and checked by reading, not compiled, not run on hardware
//
#include <hip/hip_runtime.h>
#include <hip/hip_bf16.h>
#include <math.h>

#define B_    2
#define C_    512
#define HW_   4096
#define NH_   8
#define HD_   64     // head dim
#define NG_   32     // groups
#define CPG_  16     // channels per group
#define OQKV_ 1536

typedef __attribute__((ext_vector_type(16))) __bf16 v16bf;
typedef __attribute__((ext_vector_type(8)))  float  v8f;
typedef unsigned int  u32;
typedef unsigned short u16;
typedef __attribute__((ext_vector_type(4))) u32   v4u;
typedef __attribute__((ext_vector_type(4))) float v4f;

// float -> bf16 (round to nearest even, ignoring NaN payload edge cases)
__device__ __forceinline__ u16 f2bf(float f) {
  u32 u = __float_as_uint(f);
  u32 r = u + 0x7FFFu + ((u >> 16) & 1u);
  return (u16)(r >> 16);
}

__device__ __forceinline__ v8f wmma_bf(v16bf a, v16bf b, v8f c) {
  return __builtin_amdgcn_wmma_f32_16x16x32_bf16(false, a, false, b, (short)0, c, false, false);
}

union Frag { v4u q[2]; v16bf v; };

// ---------------------------------------------------------------------------
// GroupNorm statistics: one block per (b, group); 16 ch x 4096 px reduction
// ---------------------------------------------------------------------------
__global__ void gn_stats_kernel(const float* __restrict__ x, float* __restrict__ stats) {
  int bg = blockIdx.x;                 // 0..63
  int b = bg / NG_, g = bg % NG_;
  const float* base = x + ((size_t)(b * C_ + g * CPG_)) * HW_;
  const int N = CPG_ * HW_;
  float s = 0.f, ss = 0.f;
  for (int t = threadIdx.x; t < N; t += blockDim.x) {
    float v = base[t];
    s += v; ss += v * v;
  }
  for (int msk = 16; msk >= 1; msk >>= 1) {
    s  += __shfl_xor(s,  msk, 32);
    ss += __shfl_xor(ss, msk, 32);
  }
  __shared__ float rs[8], rss[8];
  int wave = threadIdx.x >> 5;
  if ((threadIdx.x & 31) == 0) { rs[wave] = s; rss[wave] = ss; }
  __syncthreads();
  if (threadIdx.x == 0) {
    float S = 0.f, SS = 0.f;
    for (int i = 0; i < 8; ++i) { S += rs[i]; SS += rss[i]; }
    float mean = S / (float)N;
    float var  = SS / (float)N - mean * mean;
    stats[bg * 2]     = mean;
    stats[bg * 2 + 1] = rsqrtf(var + 1e-5f);
  }
}

// ---------------------------------------------------------------------------
// GroupNorm apply: x[b][c][hw] -> hN[b][hw][c] (bf16, transposed for A-frags)
// ---------------------------------------------------------------------------
__global__ void gn_apply_kernel(const float* __restrict__ x, const float* __restrict__ stats,
                                const float* __restrict__ gw, const float* __restrict__ gb,
                                u16* __restrict__ hN) {
  int idx = blockIdx.x * blockDim.x + threadIdx.x;   // over B*HW*C, c fastest
  if (idx >= B_ * HW_ * C_) return;
  int c  = idx % C_;
  int hw = (idx / C_) % HW_;
  int b  = idx / (C_ * HW_);
  int g  = c / CPG_;
  float mean = stats[(b * NG_ + g) * 2];
  float rstd = stats[(b * NG_ + g) * 2 + 1];
  float v = x[((size_t)(b * C_ + c)) * HW_ + hw];
  v = (v - mean) * rstd * gw[c] + gb[c];
  hN[idx] = f2bf(v);
}

// ---------------------------------------------------------------------------
// Pack weight w[o][c] (f32) -> wp[kb][o][kk] (bf16) for WMMA B-fragments
// ---------------------------------------------------------------------------
__global__ void pack_w_kernel(const float* __restrict__ w, u16* __restrict__ wp, int O, int C) {
  int idx = blockIdx.x * blockDim.x + threadIdx.x;   // over O*C
  if (idx >= O * C) return;
  int kk = idx & 31;
  int o  = (idx >> 5) % O;
  int kb = idx / (32 * O);
  wp[idx] = f2bf(w[(size_t)o * C + kb * 32 + kk]);
}

// ---------------------------------------------------------------------------
// QKV GEMM: qkv[b][o][i] = sum_c w[o][c] * h[b][i][c]
// Per wave: 32 rows (2 M-tiles) x 64 out-channels (4 N-tiles) = 8 WMMA / K-step,
// all 4 B-fragments issued before the WMMA burst (amortized over 2 A tiles).
// Per-head channel blocks of 192: [0,64)=Q -> qB[bh][i][d] (scaled),
// [64,128)=K -> kB[bh][i][d] (scaled), [128,192)=V -> vT[bh][c][j] (128b stores)
// ---------------------------------------------------------------------------
__global__ void __launch_bounds__(128)
qkv_kernel(const u16* __restrict__ hN, const u16* __restrict__ wp,
           u16* __restrict__ qB, u16* __restrict__ kB, u16* __restrict__ vT) {
  const int lane = threadIdx.x & 31, wave = threadIdx.x >> 5;
  const int m = lane & 15, half = lane >> 4;
  const int b  = blockIdx.z;
  const int i0 = blockIdx.x * 128 + wave * 32;
  const int o0 = blockIdx.y * 64;

  v8f acc[4][2] = {};    // [nt][mt]
  const u16* aRow0 = hN + ((size_t)(b * HW_) + i0 + m) * C_;
  const u16* aRow1 = aRow0 + 16 * C_;
  for (int kb = 0; kb < C_ / 32; ++kb) {
    Frag A[2];
    const u16* ap0 = aRow0 + kb * 32 + half * 8;
    const u16* ap1 = aRow1 + kb * 32 + half * 8;
    A[0].q[0] = *(const v4u*)ap0;
    A[0].q[1] = *(const v4u*)(ap0 + 16);
    A[1].q[0] = *(const v4u*)ap1;
    A[1].q[1] = *(const v4u*)(ap1 + 16);
    Frag Bf[4];
#pragma unroll
    for (int nt = 0; nt < 4; ++nt) {
      const u16* bp = wp + ((size_t)kb * OQKV_ + o0 + nt * 16 + m) * 32 + half * 8;
      Bf[nt].q[0] = *(const v4u*)bp;
      Bf[nt].q[1] = *(const v4u*)(bp + 16);
    }
#pragma unroll
    for (int nt = 0; nt < 4; ++nt) {
      acc[nt][0] = wmma_bf(A[0].v, Bf[nt].v, acc[nt][0]);
      acc[nt][1] = wmma_bf(A[1].v, Bf[nt].v, acc[nt][1]);
    }
  }

  const float scale = 0.35355339059327373f;  // (C/NH)^-0.25 = 64^-0.25
#pragma unroll
  for (int mt = 0; mt < 2; ++mt) {
    const int ibase = i0 + mt * 16;
#pragma unroll
    for (int nt = 0; nt < 4; ++nt) {
      int o    = o0 + nt * 16 + m;
      int head = o / 192;               // uniform within a 16-wide tile
      int r    = o % 192;
      int bh   = b * NH_ + head;
      int seg  = r / 64, rl = r % 64;   // seg uniform within tile
      if (seg == 0) {                   // Q
        u16* dst = qB + ((size_t)bh * HW_) * HD_ + rl;
#pragma unroll
        for (int r8 = 0; r8 < 8; ++r8) {
          int i = ibase + half * 8 + r8;
          dst[(size_t)i * HD_] = f2bf(acc[nt][mt][r8] * scale);
        }
      } else if (seg == 1) {            // K
        u16* dst = kB + ((size_t)bh * HW_) * HD_ + rl;
#pragma unroll
        for (int r8 = 0; r8 < 8; ++r8) {
          int i = ibase + half * 8 + r8;
          dst[(size_t)i * HD_] = f2bf(acc[nt][mt][r8] * scale);
        }
      } else {                          // V transposed: 8 consecutive j per lane
        union { v4u q; u16 h[8]; } P;
#pragma unroll
        for (int r8 = 0; r8 < 8; ++r8) P.h[r8] = f2bf(acc[nt][mt][r8]);
        u16* dst = vT + ((size_t)(bh * HD_) + rl) * HW_ + ibase + half * 8;
        *(v4u*)dst = P.q;
      }
    }
  }
}

// ---------------------------------------------------------------------------
// Flash attention: per (bh, 64-row i-block); 4 waves x one 16-row tile each.
// Per j-step the block stages the shared K (64x64) and V (64x64) tiles through
// LDS once (coalesced 128-bit global loads + ds_store_b128), then every wave
// builds its WMMA fragments from LDS. Online softmax uses 16-lane reductions
// aligned with the C/D VGPR layout; P is re-fragmented via a wave-local LDS
// staging tile.
// ---------------------------------------------------------------------------
__global__ void __launch_bounds__(128)
attn_kernel(const u16* __restrict__ qB, const u16* __restrict__ kB,
            const u16* __restrict__ vT, u16* __restrict__ hO) {
  const int lane = threadIdx.x & 31, wave = threadIdx.x >> 5;
  const int m = lane & 15, half = lane >> 4;
  const int bh = blockIdx.y;
  const int b = bh / NH_, head = bh % NH_;
  const int i0 = blockIdx.x * 64 + wave * 16;

  __shared__ u16 kT[64][72];       // K tile [j_local][d], rows 16B-aligned (144B)
  __shared__ u16 vS[64][72];       // V tile [c][j_local]
  __shared__ u16 pst[4][16][80];   // per-wave P staging

  // staging assignment: each thread copies half a 64-element row (32 u16)
  const int srow  = threadIdx.x >> 1;         // 0..63
  const int spart = (threadIdx.x & 1) * 32;   // 0 or 32

  Frag qf[2];
  {
    const u16* qp = qB + ((size_t)(bh * HW_) + i0 + m) * HD_;
#pragma unroll
    for (int kb = 0; kb < 2; ++kb) {
      qf[kb].q[0] = *(const v4u*)(qp + kb * 32 + half * 8);
      qf[kb].q[1] = *(const v4u*)(qp + kb * 32 + half * 8 + 16);
    }
  }

  float mrow[8], lrow[8];
#pragma unroll
  for (int r = 0; r < 8; ++r) { mrow[r] = -3.0e38f; lrow[r] = 0.f; }
  v8f oacc[4] = {};

  const u16* kgBase = kB + ((size_t)(bh * HW_) + srow) * HD_ + spart;
  const u16* vgBase = vT + ((size_t)(bh * HD_) + srow) * HW_ + spart;

  for (int jb = 0; jb < HW_ / 64; ++jb) {
    const int j0 = jb * 64;

    // ---- stage K and V tiles into LDS (global_load_b128 + ds_store_b128) ----
    {
      const u16* kg = kgBase + (size_t)j0 * HD_;
      const u16* vg = vgBase + j0;
#pragma unroll
      for (int q = 0; q < 4; ++q) {
        *(v4u*)&kT[srow][spart + q * 8] = *(const v4u*)(kg + q * 8);
        *(v4u*)&vS[srow][spart + q * 8] = *(const v4u*)(vg + q * 8);
      }
    }
    __syncthreads();

    // ---- S = Q K^T ----
    v8f s[4] = {};
#pragma unroll
    for (int kb = 0; kb < 2; ++kb) {
      Frag bf[4];
#pragma unroll
      for (int nt = 0; nt < 4; ++nt) {
        const u16* kp = &kT[nt * 16 + m][kb * 32 + half * 8];
        bf[nt].q[0] = *(const v4u*)kp;
        bf[nt].q[1] = *(const v4u*)(kp + 16);
      }
#pragma unroll
      for (int nt = 0; nt < 4; ++nt)
        s[nt] = wmma_bf(qf[kb].v, bf[nt].v, s[nt]);
    }

    // ---- online softmax: row r (M = r + half*8) lives in VGPR r ----
    float p[4][8];
#pragma unroll
    for (int r = 0; r < 8; ++r) {
      float rmax = s[0][r];
#pragma unroll
      for (int nt = 1; nt < 4; ++nt) rmax = fmaxf(rmax, s[nt][r]);
#pragma unroll
      for (int msk = 8; msk >= 1; msk >>= 1) rmax = fmaxf(rmax, __shfl_xor(rmax, msk, 32));
      float mnew  = fmaxf(mrow[r], rmax);
      float alpha = __expf(mrow[r] - mnew);
      float rsum = 0.f;
#pragma unroll
      for (int nt = 0; nt < 4; ++nt) {
        float e = __expf(s[nt][r] - mnew);
        p[nt][r] = e; rsum += e;
      }
#pragma unroll
      for (int msk = 8; msk >= 1; msk >>= 1) rsum += __shfl_xor(rsum, msk, 32);
      lrow[r] = lrow[r] * alpha + rsum;
      mrow[r] = mnew;
#pragma unroll
      for (int nt = 0; nt < 4; ++nt) oacc[nt][r] *= alpha;
    }

    // ---- stage P (bf16): D-layout -> A-fragment layout (wave-local) ----
#pragma unroll
    for (int nt = 0; nt < 4; ++nt)
#pragma unroll
      for (int r = 0; r < 8; ++r)
        pst[wave][half * 8 + r][nt * 16 + m] = f2bf(p[nt][r]);
    __builtin_amdgcn_wave_barrier();

    // ---- O += P V ----
#pragma unroll
    for (int kb = 0; kb < 2; ++kb) {
      Frag af;
      const u16* pp = &pst[wave][m][kb * 32 + half * 8];
      af.q[0] = *(const v4u*)pp;
      af.q[1] = *(const v4u*)(pp + 16);
      Frag bf[4];
#pragma unroll
      for (int nt = 0; nt < 4; ++nt) {
        const u16* vp = &vS[nt * 16 + m][kb * 32 + half * 8];
        bf[nt].q[0] = *(const v4u*)vp;
        bf[nt].q[1] = *(const v4u*)(vp + 16);
      }
#pragma unroll
      for (int nt = 0; nt < 4; ++nt)
        oacc[nt] = wmma_bf(af.v, bf[nt].v, oacc[nt]);
    }
    __syncthreads();   // protect kT/vS before next iteration's staging
  }

  // epilogue: O /= l, write hO[b][i][head*64 + c] (bf16, A-frag layout for proj)
#pragma unroll
  for (int r = 0; r < 8; ++r) {
    float inv = 1.0f / lrow[r];
    int i = i0 + half * 8 + r;
    u16* dst = hO + ((size_t)(b * HW_) + i) * C_ + head * HD_;
#pragma unroll
    for (int nt = 0; nt < 4; ++nt)
      dst[nt * 16 + m] = f2bf(oacc[nt][r] * inv);
  }
}

// ---------------------------------------------------------------------------
// Proj GEMM + bias + residual: out[b][o][i] = x + b[o] + sum_c w[o][c]*hO[b][i][c]
// Same 32-row x 64-col per-wave tiling as the QKV GEMM.
// ---------------------------------------------------------------------------
__global__ void __launch_bounds__(128)
proj_kernel(const u16* __restrict__ hO, const u16* __restrict__ wp,
            const float* __restrict__ pb, const float* __restrict__ x,
            float* __restrict__ out) {
  const int lane = threadIdx.x & 31, wave = threadIdx.x >> 5;
  const int m = lane & 15, half = lane >> 4;
  const int b  = blockIdx.z;
  const int i0 = blockIdx.x * 128 + wave * 32;
  const int o0 = blockIdx.y * 64;

  v8f acc[4][2] = {};
  const u16* aRow0 = hO + ((size_t)(b * HW_) + i0 + m) * C_;
  const u16* aRow1 = aRow0 + 16 * C_;
  for (int kb = 0; kb < C_ / 32; ++kb) {
    Frag A[2];
    const u16* ap0 = aRow0 + kb * 32 + half * 8;
    const u16* ap1 = aRow1 + kb * 32 + half * 8;
    A[0].q[0] = *(const v4u*)ap0;
    A[0].q[1] = *(const v4u*)(ap0 + 16);
    A[1].q[0] = *(const v4u*)ap1;
    A[1].q[1] = *(const v4u*)(ap1 + 16);
    Frag Bf[4];
#pragma unroll
    for (int nt = 0; nt < 4; ++nt) {
      const u16* bp = wp + ((size_t)kb * C_ + o0 + nt * 16 + m) * 32 + half * 8;
      Bf[nt].q[0] = *(const v4u*)bp;
      Bf[nt].q[1] = *(const v4u*)(bp + 16);
    }
#pragma unroll
    for (int nt = 0; nt < 4; ++nt) {
      acc[nt][0] = wmma_bf(A[0].v, Bf[nt].v, acc[nt][0]);
      acc[nt][1] = wmma_bf(A[1].v, Bf[nt].v, acc[nt][1]);
    }
  }

#pragma unroll
  for (int mt = 0; mt < 2; ++mt) {
    const int ibase = i0 + mt * 16;
#pragma unroll
    for (int nt = 0; nt < 4; ++nt) {
      int o = o0 + nt * 16 + m;
      float bias = pb[o];
      size_t base = ((size_t)(b * C_) + o) * HW_ + ibase + half * 8;  // 8 consecutive i
      v4f x0 = *(const v4f*)(x + base);
      v4f x1 = *(const v4f*)(x + base + 4);
      v4f r0, r1;
      r0.x = x0.x + bias + acc[nt][mt][0];
      r0.y = x0.y + bias + acc[nt][mt][1];
      r0.z = x0.z + bias + acc[nt][mt][2];
      r0.w = x0.w + bias + acc[nt][mt][3];
      r1.x = x1.x + bias + acc[nt][mt][4];
      r1.y = x1.y + bias + acc[nt][mt][5];
      r1.z = x1.z + bias + acc[nt][mt][6];
      r1.w = x1.w + bias + acc[nt][mt][7];
      *(v4f*)(out + base)     = r0;
      *(v4f*)(out + base + 4) = r1;
    }
  }
}

// ---------------------------------------------------------------------------
extern "C" void kernel_launch(void* const* d_in, const int* in_sizes, int n_in,
                              void* d_out, int out_size, void* d_ws, size_t ws_size,
                              hipStream_t stream) {
  const float* x     = (const float*)d_in[0];
  const float* qkvw  = (const float*)d_in[1];
  const float* projw = (const float*)d_in[2];
  const float* projb = (const float*)d_in[3];
  const float* gnw   = (const float*)d_in[4];
  const float* gnb   = (const float*)d_in[5];
  float* out = (float*)d_out;

  char* p = (char*)d_ws;
  auto alloc = [&](size_t bytes) -> char* {
    char* r = p;
    p += (bytes + 255) & ~(size_t)255;
    return r;
  };
  float* stats = (float*)alloc((size_t)B_ * NG_ * 2 * sizeof(float));
  u16* hN  = (u16*)alloc((size_t)B_ * HW_ * C_ * 2);           // 8 MB
  u16* wqp = (u16*)alloc((size_t)OQKV_ * C_ * 2);              // 1.5 MB
  u16* wpp = (u16*)alloc((size_t)C_ * C_ * 2);                 // 0.5 MB
  u16* qB  = (u16*)alloc((size_t)B_ * NH_ * HW_ * HD_ * 2);    // 8 MB
  u16* kB  = (u16*)alloc((size_t)B_ * NH_ * HW_ * HD_ * 2);    // 8 MB
  u16* vT  = (u16*)alloc((size_t)B_ * NH_ * HW_ * HD_ * 2);    // 8 MB
  u16* hO  = (u16*)alloc((size_t)B_ * HW_ * C_ * 2);           // 8 MB

  gn_stats_kernel<<<B_ * NG_, 256, 0, stream>>>(x, stats);
  gn_apply_kernel<<<(B_ * HW_ * C_) / 256, 256, 0, stream>>>(x, stats, gnw, gnb, hN);
  pack_w_kernel<<<(OQKV_ * C_) / 256, 256, 0, stream>>>(qkvw, wqp, OQKV_, C_);
  pack_w_kernel<<<(C_ * C_) / 256, 256, 0, stream>>>(projw, wpp, C_, C_);
  qkv_kernel<<<dim3(HW_ / 128, OQKV_ / 64, B_), 128, 0, stream>>>(hN, wqp, qB, kB, vT);
  attn_kernel<<<dim3(HW_ / 64, B_ * NH_), 128, 0, stream>>>(qB, kB, vT, hO);
  proj_kernel<<<dim3(HW_ / 128, C_ / 64, B_), 128, 0, stream>>>(hO, wpp, projb, x, out);
}
